// LARoPECrossAttention_55327768708520
// MI455X (gfx1250) — compile-verified, matrix operand order checked by reading, not run on hardware
//
#include <hip/hip_runtime.h>
#include <hip/hip_bf16.h>
#include <math.h>

typedef __attribute__((ext_vector_type(16))) __bf16 v16bf;
typedef __attribute__((ext_vector_type(8)))  __bf16 v8bf;
typedef __attribute__((ext_vector_type(8)))  float  v8f;

static __device__ __forceinline__ __bf16 f2bf(float f) {
    unsigned u = __float_as_uint(f);
    unsigned r = (u + 0x7FFFu + ((u >> 16) & 1u)) >> 16;
    unsigned short s = (unsigned short)r;
    return __builtin_bit_cast(__bf16, s);
}

static __device__ __forceinline__ v16bf cat8(v8bf lo, v8bf hi) {
    return __builtin_shufflevector(lo, hi, 0,1,2,3,4,5,6,7,8,9,10,11,12,13,14,15);
}

// ---- CDNA5 async copy: global -> LDS, 128b per lane, tracked by ASYNCcnt ----
static __device__ __forceinline__ void async_b128(unsigned lds_off, const void* gptr) {
    asm volatile("global_load_async_to_lds_b128 %0, %1, off"
                 :: "v"(lds_off), "v"(gptr) : "memory");
}
static __device__ __forceinline__ void wait_async0() {
    asm volatile("s_wait_asynccnt 0x0" ::: "memory");
}
static __device__ __forceinline__ void wait_async2() {
    asm volatile("s_wait_asynccnt 0x2" ::: "memory");
}
static __device__ __forceinline__ void wait_async5() {
    asm volatile("s_wait_asynccnt 0x5" ::: "memory");
}
static __device__ __forceinline__ unsigned lds_off(const void* p) {
    return (unsigned)(unsigned long long)p;   // low 32 bits of shared ptr = LDS byte offset
}

// ---------------------------------------------------------------------------
// fp32 -> bf16 conversion (grid-stride)
// ---------------------------------------------------------------------------
__global__ void convert_f32_bf16(const float* __restrict__ in,
                                 __bf16* __restrict__ out, int n) {
    int i = blockIdx.x * blockDim.x + threadIdx.x;
    int stride = gridDim.x * blockDim.x;
    for (; i < n; i += stride) out[i] = f2bf(in[i]);
}

// fp32 [K][N] -> bf16 transposed [N][K]  (weights only; small, done once)
__global__ void convert_transpose(const float* __restrict__ in,
                                  __bf16* __restrict__ out, int K, int N) {
    int idx = blockIdx.x * blockDim.x + threadIdx.x;
    int total = K * N;
    int stride = gridDim.x * blockDim.x;
    for (; idx < total; idx += stride) {
        int k = idx / N, n = idx % N;
        out[(long)n * K + k] = f2bf(in[idx]);
    }
}

// ---------------------------------------------------------------------------
// WMMA bf16 GEMM:  C[M,N] = A[M,K] * BT[N,K]^T + bias[N]
// K and epilogue MODE are compile-time: 32-step async double-buffered pipeline
// fully unrolled (static buffer indices, no selects).
// Block: 256 threads = 8 waves (2x4), block tile 64x256, BK=32.
// Wave tile 32x64 = 2x4 WMMA 16x16 tiles (8 WMMA per 12 fragment ds_loads).
// MODE 0: fp32 out; MODE 1: Q+LARoPE (Lq=2048); MODE 2: K+LARoPE / V-transpose
// ---------------------------------------------------------------------------
#define BM 64
#define BN 256
#define BK 32
#define LDP 40   // padded LDS row (80B stride: 16B-aligned, conflict-free)

template <int K, int MODE>
__global__ __launch_bounds__(256) void gemm_bf16(
        const __bf16* __restrict__ A, const __bf16* __restrict__ BT,
        const float* __restrict__ bias,
        float* __restrict__ outF, __bf16* __restrict__ out0,
        __bf16* __restrict__ out1, int N) {
    __shared__ __bf16 as[2][BM][LDP];
    __shared__ __bf16 bs[2][BN][LDP];

    const int tid  = threadIdx.x;
    const int lane = tid & 31;
    const int wave = tid >> 5;
    const int wm   = wave >> 2;          // 0..1
    const int wn   = wave & 3;           // 0..3
    const int bm   = blockIdx.y * BM;
    const int bn   = blockIdx.x * BN;
    const int l16  = lane & 15;
    const int half8 = (lane < 16) ? 0 : 8;

    v8f acc[2][4] = {};

    // async chunk assignment: A 256 chunks (1/thread), B 1024 chunks (4/thread)
    const int crow = tid >> 2;             // 0..63
    const int cch  = (tid & 3) * 8;        // 0,8,16,24
    const __bf16* aptr = A + (long)(bm + crow) * K + cch;
    const __bf16* bptr[4];
    unsigned la[2], lb[4][2];
    #pragma unroll
    for (int r = 0; r < 4; ++r) {
        bptr[r] = BT + (long)(bn + r * 64 + crow) * K + cch;
        lb[r][0] = lds_off(&bs[0][r * 64 + crow][cch]);
        lb[r][1] = lds_off(&bs[1][r * 64 + crow][cch]);
    }
    la[0] = lds_off(&as[0][crow][cch]);
    la[1] = lds_off(&as[1][crow][cch]);

    // prime stage 0
    async_b128(la[0], aptr);
    #pragma unroll
    for (int r = 0; r < 4; ++r) async_b128(lb[r][0], bptr[r]);

    constexpr int NT = K / BK;
    #pragma unroll
    for (int kt = 0; kt < NT; ++kt) {
        const int buf = kt & 1;
        if (kt + 1 < NT) {
            aptr += BK;
            async_b128(la[buf ^ 1], aptr);
            #pragma unroll
            for (int r = 0; r < 4; ++r) {
                bptr[r] += BK;
                async_b128(lb[r][buf ^ 1], bptr[r]);
            }
            wait_async5();               // current tile landed; next 5 in flight
        } else {
            wait_async0();
        }
        __syncthreads();

        #pragma unroll
        for (int mt = 0; mt < 2; ++mt) {
            int ar = wm * 32 + mt * 16 + l16;
            v16bf af = cat8(*(const v8bf*)&as[buf][ar][half8],
                            *(const v8bf*)&as[buf][ar][half8 + 16]);
            #pragma unroll
            for (int nt = 0; nt < 4; ++nt) {
                int br = wn * 64 + nt * 16 + l16;
                v16bf bfm = cat8(*(const v8bf*)&bs[buf][br][half8],
                                 *(const v8bf*)&bs[buf][br][half8 + 16]);
                acc[mt][nt] = __builtin_amdgcn_wmma_f32_16x16x32_bf16(
                    false, af, false, bfm, (short)0, acc[mt][nt], false, false);
            }
        }
        __syncthreads();
    }

    // ---- epilogue ----
    #pragma unroll
    for (int mt = 0; mt < 2; ++mt) {
        #pragma unroll
        for (int nt = 0; nt < 4; ++nt) {
            int col = bn + wn * 64 + nt * 16 + l16;
            float bv = bias[col];
            #pragma unroll
            for (int i = 0; i < 8; ++i) {
                int row = bm + wm * 32 + mt * 16 + i + half8;
                float v = acc[mt][nt][i] + bv;
                if (MODE == 0) {
                    outF[(long)row * N + col] = v;
                } else if (MODE == 1) {
                    int b = row >> 11, qpos = row & 2047;
                    int h = col >> 6,  d = col & 63;
                    float pos = 10.0f * (float)qpos * (1.0f / 2048.0f);
                    float ang = pos * __expf(-0.2878231366242557f * (float)(d >> 1));
                    float cs = __cosf(ang), sn = __sinf(ang);
                    float p = __shfl_xor(v, 1, 32);   // pair partner (col^1)
                    float o = (d & 1) ? (p * sn + v * cs) : (v * cs - p * sn);
                    out0[(((long)b * 16 + h) * 2048 + qpos) * 64 + d] = f2bf(o);
                } else {
                    int b = row >> 9, kpos = row & 511;
                    if (col < 1024) {
                        int h = col >> 6, d = col & 63;
                        float pos = 10.0f * (float)kpos * (1.0f / 512.0f);
                        float ang = pos * __expf(-0.2878231366242557f * (float)(d >> 1));
                        float cs = __cosf(ang), sn = __sinf(ang);
                        float p = __shfl_xor(v, 1, 32);
                        float o = (d & 1) ? (p * sn + v * cs) : (v * cs - p * sn);
                        out0[(((long)b * 16 + h) * 512 + kpos) * 64 + d] = f2bf(o);
                    } else {
                        int cc = col - 1024;
                        int h = cc >> 6, d = cc & 63;
                        out1[(((long)b * 16 + h) * 64 + d) * 512 + kpos] = f2bf(v);
                    }
                }
            }
        }
    }
}

// ---------------------------------------------------------------------------
// Flash attention. All 8 waves of a block share one (b,h): K/V 32-key tiles are
// async-staged into LDS once per block (double-buffered), each wave owns 16
// queries and runs online softmax on 16x16 WMMA S-tiles.
// ---------------------------------------------------------------------------
#define KPD 72   // K-tile LDS row pad (144B stride)
#define VPD 40   // V-tile LDS row pad (80B stride)

__global__ __launch_bounds__(256) void attn_kernel(
        const __bf16* __restrict__ qb, const __bf16* __restrict__ kbm,
        const __bf16* __restrict__ vt, const unsigned char* __restrict__ mask,
        __bf16* __restrict__ aob) {
    __shared__ __bf16 ks[2][32][KPD];    // 32 keys x 64 dims
    __shared__ __bf16 vs[2][64][VPD];    // 64 dims x 32 keys (from V^T)
    __shared__ __bf16 pbuf[8][16][32];   // per-wave P relayout buffer

    const int tid  = threadIdx.x;
    const int lane = tid & 31;
    const int wave = tid >> 5;
    const int w    = blockIdx.x * 8 + wave;
    const int qt   = w & 127;
    const int h    = (w >> 7) & 15;
    const int b    = w >> 11;
    const int l16  = lane & 15;
    const int half8 = (lane < 16) ? 0 : 8;

    const __bf16* kbase = kbm + (((long)b * 16 + h) * 512) * 64;
    const __bf16* vbase = vt  + (((long)b * 16 + h) * 64) * 512;
    const unsigned char* mbase = mask + b * 512;

    // async chunk assignment: K 256 chunks + V 256 chunks = 2/thread
    const int krow = tid >> 3, kch = (tid & 7) * 8;   // K: 32 rows x 8 chunks
    const int vrow = tid >> 2, vch = (tid & 3) * 8;   // V: 64 rows x 4 chunks
    const __bf16* kptr = kbase + (long)krow * 64 + kch;
    const __bf16* vptr = vbase + (long)vrow * 512 + vch;
    unsigned lk[2], lv[2];
    lk[0] = lds_off(&ks[0][krow][kch]); lk[1] = lds_off(&ks[1][krow][kch]);
    lv[0] = lds_off(&vs[0][vrow][vch]); lv[1] = lds_off(&vs[1][vrow][vch]);

    // Q fragments (16x64), kept in registers
    const __bf16* qp = qb + (((long)b * 16 + h) * 2048 + qt * 16 + l16) * 64;
    v16bf qf[2];
    #pragma unroll
    for (int s = 0; s < 2; ++s)
        qf[s] = cat8(*(const v8bf*)&qp[s * 32 + half8],
                     *(const v8bf*)&qp[s * 32 + half8 + 16]);

    v8f o[4] = {};
    float m[8], l[8];
    #pragma unroll
    for (int i = 0; i < 8; ++i) { m[i] = -3.0e38f; l[i] = 0.0f; }

    // prime stage 0
    async_b128(lk[0], kptr);
    async_b128(lv[0], vptr);

    int buf = 0;
    for (int kb0 = 0; kb0 < 512; kb0 += 32) {
        if (kb0 + 32 < 512) {
            kptr += 32 * 64;   // next 32 keys (row-major K)
            vptr += 32;        // next 32 keys (V^T columns)
            async_b128(lk[buf ^ 1], kptr);
            async_b128(lv[buf ^ 1], vptr);
            wait_async2();
        } else {
            wait_async0();
        }
        __syncthreads();

        // S = Q*K^T for 32 keys -> two 16x16 fp32 tiles
        v8f sacc[2] = {};
        #pragma unroll
        for (int nt = 0; nt < 2; ++nt) {
            #pragma unroll
            for (int s = 0; s < 2; ++s) {
                int kr = nt * 16 + l16;
                v16bf kf = cat8(*(const v8bf*)&ks[buf][kr][s * 32 + half8],
                                *(const v8bf*)&ks[buf][kr][s * 32 + half8 + 16]);
                sacc[nt] = __builtin_amdgcn_wmma_f32_16x16x32_bf16(
                    false, qf[s], false, kf, (short)0, sacc[nt], false, false);
            }
        }
        float mk0 = mbase[kb0 + l16]      ? 0.0f : -3.0e38f;
        float mk1 = mbase[kb0 + 16 + l16] ? 0.0f : -3.0e38f;

        // online softmax; row i lives in C-VGPR i across a 16-lane half-group
        float pr0[8], pr1[8];
        #pragma unroll
        for (int i = 0; i < 8; ++i) {
            float s0 = sacc[0][i] * 0.125f + mk0;
            float s1 = sacc[1][i] * 0.125f + mk1;
            float t = fmaxf(s0, s1);
            #pragma unroll
            for (int off = 1; off < 16; off <<= 1) t = fmaxf(t, __shfl_xor(t, off, 32));
            float mn = fmaxf(m[i], t);
            float alpha = __expf(m[i] - mn);
            float p0 = __expf(s0 - mn);
            float p1 = __expf(s1 - mn);
            float rs = p0 + p1;
            #pragma unroll
            for (int off = 1; off < 16; off <<= 1) rs += __shfl_xor(rs, off, 32);
            l[i] = l[i] * alpha + rs;
            m[i] = mn;
            #pragma unroll
            for (int t4 = 0; t4 < 4; ++t4) o[t4][i] *= alpha;
            pr0[i] = p0; pr1[i] = p1;
        }

        // P (C-tile layout) -> LDS -> A-fragment layout
        #pragma unroll
        for (int i = 0; i < 8; ++i) {
            pbuf[wave][i + half8][l16]      = f2bf(pr0[i]);
            pbuf[wave][i + half8][16 + l16] = f2bf(pr1[i]);
        }
        v16bf pf = cat8(*(const v8bf*)&pbuf[wave][l16][half8],
                        *(const v8bf*)&pbuf[wave][l16][half8 + 16]);

        // O += P * V
        #pragma unroll
        for (int t4 = 0; t4 < 4; ++t4) {
            int vr = t4 * 16 + l16;
            v16bf vf = cat8(*(const v8bf*)&vs[buf][vr][half8],
                            *(const v8bf*)&vs[buf][vr][half8 + 16]);
            o[t4] = __builtin_amdgcn_wmma_f32_16x16x32_bf16(
                false, pf, false, vf, (short)0, o[t4], false, false);
        }
        __syncthreads();
        buf ^= 1;
    }

    // normalize and write merged-head bf16 output [8,2048,1024]
    #pragma unroll
    for (int t4 = 0; t4 < 4; ++t4) {
        #pragma unroll
        for (int i = 0; i < 8; ++i) {
            int qrow = qt * 16 + i + half8;
            int d = t4 * 16 + l16;
            float ov = o[t4][i] / l[i];
            aob[((long)b * 2048 + qrow) * 1024 + h * 64 + d] = f2bf(ov);
        }
    }
}

// ---------------------------------------------------------------------------
// Host launcher
// ---------------------------------------------------------------------------
extern "C" void kernel_launch(void* const* d_in, const int* in_sizes, int n_in,
                              void* d_out, int out_size, void* d_ws, size_t ws_size,
                              hipStream_t stream) {
    const float* x    = (const float*)d_in[0];
    const float* ctx  = (const float*)d_in[1];
    const unsigned char* cmask = (const unsigned char*)d_in[2];
    const float* Wq   = (const float*)d_in[3];
    const float* bq   = (const float*)d_in[4];
    const float* Wkv  = (const float*)d_in[5];
    const float* bkv  = (const float*)d_in[6];
    const float* Wo   = (const float*)d_in[7];
    const float* bo   = (const float*)d_in[8];
    float* out = (float*)d_out;

    char* ws = (char*)d_ws;
    __bf16* xb   = (__bf16*)(ws);                   // x bf16          32 MB
    __bf16* cb   = (__bf16*)(ws + 33554432);        // context bf16     8 MB
    __bf16* wqb  = (__bf16*)(ws + 41943040);        // Wq^T bf16        2 MB
    __bf16* wkvb = (__bf16*)(ws + 44040192);        // Wkv^T bf16       4 MB
    __bf16* wob  = (__bf16*)(ws + 48234496);        // Wo^T bf16        2 MB
    __bf16* qb   = (__bf16*)(ws + 50331648);        // q roped         32 MB
    __bf16* kbm  = (__bf16*)(ws + 83886080);        // k roped          8 MB
    __bf16* vt   = (__bf16*)(ws + 92274688);        // v transposed     8 MB
    __bf16* aob  = (__bf16*)(ws + 100663296);       // attn out        32 MB
    // total workspace: 134217728 bytes (128 MB)

    convert_f32_bf16<<<512, 256, 0, stream>>>(x,   xb, 16777216);
    convert_f32_bf16<<<256, 256, 0, stream>>>(ctx, cb, 4194304);
    convert_transpose<<<256, 256, 0, stream>>>(Wq,  wqb,  1024, 1024);
    convert_transpose<<<512, 256, 0, stream>>>(Wkv, wkvb, 1024, 2048);
    convert_transpose<<<256, 256, 0, stream>>>(Wo,  wob,  1024, 1024);

    dim3 blk(256);
    // Q = x*Wq + bq, fused RoPE  (M=16384, N=1024)
    gemm_bf16<1024, 1><<<dim3(1024 / BN, 16384 / BM), blk, 0, stream>>>(
        xb, wqb, bq, nullptr, qb, nullptr, 1024);
    // KV = ctx*Wkv + bkv, fused RoPE(K) + V-transpose  (M=4096, N=2048)
    gemm_bf16<1024, 2><<<dim3(2048 / BN, 4096 / BM), blk, 0, stream>>>(
        cb, wkvb, bkv, nullptr, kbm, vt, 2048);
    // flash attention: 16384 wave-tiles, 8 waves/block
    attn_kernel<<<2048, blk, 0, stream>>>(qb, kbm, vt, cmask, aob);
    // out = attn*Wo + bo  (fp32 output)
    gemm_bf16<1024, 0><<<dim3(1024 / BN, 16384 / BM), blk, 0, stream>>>(
        aob, wob, bo, out, nullptr, nullptr, 1024);
}
